// VQLayer_37039797961385
// MI455X (gfx1250) — matrix operand order repaired and checked by reading, not compiled, not run on hardware
//
#include <hip/hip_runtime.h>
#include <hip/hip_bf16.h>

// ---------------------------------------------------------------------------
// VQ layer fused kernel for MI455X (gfx1250, wave32, WMMA).
//
//   dists = |x|^2 + |p|^2 - 2 x.p^T   -> only t = |p|^2 - 2 x.p matters
//   argmin / softmax(-t) per row; soft[k] = mean_n softmax; entropy(soft)
//   vq_loss = 1.25 * mean((q - x)^2) + 0.01 * entropy
//   output  = gathered prototype rows (forward of straight-through)
//
// GEMM: bf16 WMMA 16x16x32 with hi/lo fp32 split (3 products). Proto k-slices
// are staged to LDS with CDNA5 async direct-to-LDS copies (ASYNCcnt) in a
// ping-pong: copies for chunk i+1 are in flight while chunk i's 24 WMMAs run.
// ---------------------------------------------------------------------------

typedef __attribute__((ext_vector_type(16))) __bf16 v16bf;
typedef __attribute__((ext_vector_type(8)))  float  v8f;

#define N_TOK   65536
#define DIM     512
#define NPROTO  1024
#define BM      64          // latent rows per workgroup
#define KSLICE  32          // k-step per WMMA
#define THREADS 512         // 16 waves: 4 row-groups x 4 col-groups
#define KITERS  (DIM / KSLICE)
#define HCOLS   512         // columns per ping-pong chunk

// dynamic LDS layout (bytes)
#define LDS_BYTES 274192

// async 16B copy: global -> LDS, offset applies to BOTH addresses (ISA 8 §4.4)
#define ASYNC_CP16(ldsOff, gAddr, OFF)                                        \
    asm volatile("global_load_async_to_lds_b128 %0, %1, off offset:" #OFF     \
                 :: "v"(ldsOff), "v"(gAddr) : "memory")
#define WAIT_ASYNC() asm volatile("s_wait_asynccnt 0x0" ::: "memory")

__device__ __forceinline__ v16bf load_frag_a(const __bf16* base) {
    v16bf v;
    reinterpret_cast<uint4*>(&v)[0] = *reinterpret_cast<const uint4*>(base);
    reinterpret_cast<uint4*>(&v)[1] = *reinterpret_cast<const uint4*>(base + 8);
    return v;
}
__device__ __forceinline__ v16bf load_frag_b(const __bf16* base) {
    v16bf v;
    reinterpret_cast<uint4*>(&v)[0] = *reinterpret_cast<const uint4*>(base);
    reinterpret_cast<uint4*>(&v)[1] = *reinterpret_cast<const uint4*>(base + 8);
    return v;
}

// ---------------------------------------------------------------------------
// Kernel A: split prototypes into bf16 hi/lo planes, compute |p|^2,
// zero global accumulators. One block per prototype row.
// ---------------------------------------------------------------------------
__global__ void vq_prep(const float* __restrict__ protos,
                        __bf16* __restrict__ pHi, __bf16* __restrict__ pLo,
                        float* __restrict__ pn2,
                        float* __restrict__ gSoft, float* __restrict__ gMse) {
    __shared__ float red[256];
    const int row = blockIdx.x;
    float acc = 0.f;
    for (int k = threadIdx.x; k < DIM; k += 256) {
        float f = protos[(size_t)row * DIM + k];
        __bf16 h = (__bf16)f;
        pHi[(size_t)row * DIM + k] = h;
        pLo[(size_t)row * DIM + k] = (__bf16)(f - (float)h);
        acc += f * f;
    }
    red[threadIdx.x] = acc;
    __syncthreads();
    for (int s = 128; s > 0; s >>= 1) {
        if (threadIdx.x < s) red[threadIdx.x] += red[threadIdx.x + s];
        __syncthreads();
    }
    if (threadIdx.x == 0) {
        pn2[row]   = red[0];
        gSoft[row] = 0.f;
        if (row == 0) gMse[0] = 0.f;
    }
}

// ---------------------------------------------------------------------------
// Kernel B: fused dists-GEMM + argmin + softmax-accumulate + gather + MSE.
// 1024 workgroups x 512 threads; each WG: 64 rows x all 1024 protos.
// ---------------------------------------------------------------------------
__launch_bounds__(THREADS)
__global__ void vq_main(const float* __restrict__ latents,
                        const float* __restrict__ protos,
                        const __bf16* __restrict__ pHi,
                        const __bf16* __restrict__ pLo,
                        const float* __restrict__ pn2,
                        float* __restrict__ gSoft, float* __restrict__ gMse,
                        float* __restrict__ outQ) {
    extern __shared__ __align__(16) char smem[];
    __bf16* sAhi = reinterpret_cast<__bf16*>(smem);          // [64][512]
    __bf16* sAlo = sAhi + BM * DIM;
    __bf16* sBh0 = sAlo + BM * DIM;                          // [512][32] buf0 hi
    __bf16* sBh1 = sBh0 + HCOLS * KSLICE;                    // buf1 hi
    __bf16* sBl0 = sBh1 + HCOLS * KSLICE;                    // buf0 lo
    __bf16* sBl1 = sBl0 + HCOLS * KSLICE;                    // buf1 lo
    float*  sPn2 = reinterpret_cast<float*>(smem + 262144);  // [1024]
    float*  sSoft   = sPn2 + NPROTO;                         // [1024]
    float*  sRowMin = sSoft + NPROTO;                        // [4][64]
    float*  sRowZ   = sRowMin + 4 * BM;                      // [4][64]
    float*  sRowM   = sRowZ + 4 * BM;                        // [64]
    float*  sRowInvZ= sRowM + BM;                            // [64]
    int*    sRowArg = reinterpret_cast<int*>(sRowInvZ + BM); // [4][64]
    int*    sRowIdx = sRowArg + 4 * BM;                      // [64]
    float*  sMse    = reinterpret_cast<float*>(sRowIdx + BM);

    const int tid  = threadIdx.x;
    const int wave = tid >> 5;
    const int lane = tid & 31;
    const int wr = wave >> 2;          // row group 0..3  (rows wr*16..+16)
    const int wc = wave & 3;           // col group 0..3  (128 cols per chunk)
    const int laneHalf = lane & 15;
    const int hh = lane >> 4;          // 0: lanes 0-15, 1: lanes 16-31
    const int n0 = blockIdx.x * BM;

    // ---- load + split latent tile, prime soft/pn2 -------------------------
    for (int i = tid; i < BM * DIM; i += THREADS) {
        float f = latents[(size_t)n0 * DIM + i];
        __bf16 h = (__bf16)f;
        sAhi[i] = h;
        sAlo[i] = (__bf16)(f - (float)h);
    }
    for (int i = tid; i < NPROTO; i += THREADS) {
        sPn2[i]  = pn2[i];
        sSoft[i] = 0.f;
    }
    if (tid == 0) sMse[0] = 0.f;

    // async stage of one 512-col x 32-k chunk (1 proto row / thread, 2 planes)
    auto stage = [&](int h, int kt2) {
        const int k0 = kt2 * KSLICE;
        const int pr = h * HCOLS + tid;                 // proto row staged
        unsigned lH = (unsigned)(uintptr_t)((h ? sBh1 : sBh0) + tid * KSLICE);
        unsigned lL = (unsigned)(uintptr_t)((h ? sBl1 : sBl0) + tid * KSLICE);
        unsigned long long gH = (unsigned long long)(uintptr_t)(pHi + (size_t)pr * DIM + k0);
        unsigned long long gL = (unsigned long long)(uintptr_t)(pLo + (size_t)pr * DIM + k0);
        ASYNC_CP16(lH, gH, 0);  ASYNC_CP16(lH, gH, 16);
        ASYNC_CP16(lH, gH, 32); ASYNC_CP16(lH, gH, 48);
        ASYNC_CP16(lL, gL, 0);  ASYNC_CP16(lL, gL, 16);
        ASYNC_CP16(lL, gL, 32); ASYNC_CP16(lL, gL, 48);
        if (kt2 + 1 < KITERS) {   // warm near caches for the chunk after next
            __builtin_prefetch(pHi + (size_t)pr * DIM + k0 + KSLICE, 0, 3);
            __builtin_prefetch(pLo + (size_t)pr * DIM + k0 + KSLICE, 0, 3);
        }
    };

    // 8 tiles of one chunk, B frags software-pipelined one tile ahead
    auto computeChunk = [&](v16bf aHi, v16bf aLo,
                            const __bf16* bH, const __bf16* bL, v8f* cp) {
        const int lc0 = (wc * 128 + laneHalf) * KSLICE + hh * 16;
        v16bf fbH = load_frag_b(bH + lc0);
        v16bf fbL = load_frag_b(bL + lc0);
#pragma unroll
        for (int ct = 0; ct < 8; ++ct) {
            v16bf nH = fbH, nL = fbL;
            if (ct + 1 < 8) {
                const int lc = (wc * 128 + (ct + 1) * 16 + laneHalf) * KSLICE + hh * 16;
                nH = load_frag_b(bH + lc);
                nL = load_frag_b(bL + lc);
            }
            cp[ct] = __builtin_amdgcn_wmma_f32_16x16x32_bf16(false, aHi, false, fbH,
                                                             (short)0, cp[ct], false, false);
            cp[ct] = __builtin_amdgcn_wmma_f32_16x16x32_bf16(false, aLo, false, fbH,
                                                             (short)0, cp[ct], false, false);
            cp[ct] = __builtin_amdgcn_wmma_f32_16x16x32_bf16(false, aHi, false, fbL,
                                                             (short)0, cp[ct], false, false);
            fbH = nH; fbL = nL;
        }
    };

    // ---- GEMM: s = x . p ; C in registers (16 tiles of 16x16) -------------
    v8f c[16];
#pragma unroll
    for (int ct = 0; ct < 16; ++ct) c[ct] = v8f{};

    stage(0, 0);                       // prologue: chunk (kt=0, h=0) -> buf0
    for (int kt = 0; kt < KITERS; ++kt) {
        const int k0 = kt * KSLICE;

        // ---- chunk h=0 (cols 0..511, buf0) --------------------------------
        WAIT_ASYNC();                  // own copies for this chunk landed
        __syncthreads();               // everyone's copies + prev compute done
        stage(1, kt);                  // overlap: stage h=1 chunk into buf1

        const int   M  = wr * 16 + laneHalf;
        const int   kb = k0 + hh * 16;
        const v16bf aHi = load_frag_a(sAhi + M * DIM + kb);
        const v16bf aLo = load_frag_a(sAlo + M * DIM + kb);

        computeChunk(aHi, aLo, sBh0, sBl0, &c[0]);

        // ---- chunk h=1 (cols 512..1023, buf1) -----------------------------
        WAIT_ASYNC();
        __syncthreads();
        if (kt + 1 < KITERS) stage(0, kt + 1);   // stage next k-slice h=0

        computeChunk(aHi, aLo, sBh1, sBl1, &c[8]);
    }

    // Rows owned by this lane: rowBase + j, j = 0..7  (C VGPR j -> M = j + 8*hh)
    const int rowBase = wr * 16 + hh * 8;
    // tile tt -> column: (tt>>3)*512 + wc*128 + (tt&7)*16 + laneHalf

    // ---- Phase A: per-row argmin of t = |p|^2 - 2 s -----------------------
    float tmin[8]; int targ[8];
#pragma unroll
    for (int j = 0; j < 8; ++j) { tmin[j] = 1e30f; targ[j] = 0; }
#pragma unroll
    for (int tt = 0; tt < 16; ++tt) {
        const int col = (tt >> 3) * HCOLS + wc * 128 + (tt & 7) * 16 + laneHalf;
        const float p2 = sPn2[col];
#pragma unroll
        for (int j = 0; j < 8; ++j) {
            float t = p2 - 2.0f * c[tt][j];
            if (t < tmin[j]) { tmin[j] = t; targ[j] = col; }
        }
    }
#pragma unroll
    for (int j = 0; j < 8; ++j) {
        float m = tmin[j]; int id = targ[j];
#pragma unroll
        for (int off = 1; off < 16; off <<= 1) {
            float om = __shfl_xor(m, off, 32);
            int   oi = __shfl_xor(id, off, 32);
            if (om < m || (om == m && oi < id)) { m = om; id = oi; }
        }
        tmin[j] = m; targ[j] = id;
    }
    if (laneHalf == 0) {
#pragma unroll
        for (int j = 0; j < 8; ++j) {
            sRowMin[wc * BM + rowBase + j] = tmin[j];
            sRowArg[wc * BM + rowBase + j] = targ[j];
        }
    }
    __syncthreads();
    if (tid < BM) {
        float m = sRowMin[tid]; int id = sRowArg[tid];
        for (int g = 1; g < 4; ++g) {
            float om = sRowMin[g * BM + tid]; int oi = sRowArg[g * BM + tid];
            if (om < m || (om == m && oi < id)) { m = om; id = oi; }
        }
        sRowM[tid] = m; sRowIdx[tid] = id;
    }
    __syncthreads();

    // ---- Phase B: softmax normalizer Z = sum exp(minT - t) ----------------
    float mrow[8];
#pragma unroll
    for (int j = 0; j < 8; ++j) mrow[j] = sRowM[rowBase + j];
    float zsum[8];
#pragma unroll
    for (int j = 0; j < 8; ++j) zsum[j] = 0.f;
#pragma unroll
    for (int tt = 0; tt < 16; ++tt) {
        const int col = (tt >> 3) * HCOLS + wc * 128 + (tt & 7) * 16 + laneHalf;
        const float p2 = sPn2[col];
#pragma unroll
        for (int j = 0; j < 8; ++j)
            zsum[j] += __expf(mrow[j] - (p2 - 2.0f * c[tt][j]));
    }
#pragma unroll
    for (int j = 0; j < 8; ++j) {
#pragma unroll
        for (int off = 1; off < 16; off <<= 1) zsum[j] += __shfl_xor(zsum[j], off, 32);
    }
    if (laneHalf == 0) {
#pragma unroll
        for (int j = 0; j < 8; ++j) sRowZ[wc * BM + rowBase + j] = zsum[j];
    }
    __syncthreads();
    if (tid < BM) {
        float z = sRowZ[tid] + sRowZ[BM + tid] + sRowZ[2 * BM + tid] + sRowZ[3 * BM + tid];
        sRowInvZ[tid] = 1.0f / z;
    }
    __syncthreads();

    // ---- Phase C: accumulate soft[k] += prob over this WG's 64 rows -------
    float invz[8];
#pragma unroll
    for (int j = 0; j < 8; ++j) invz[j] = sRowInvZ[rowBase + j];
#pragma unroll
    for (int tt = 0; tt < 16; ++tt) {
        const int col = (tt >> 3) * HCOLS + wc * 128 + (tt & 7) * 16 + laneHalf;
        const float p2 = sPn2[col];
        float colSum = 0.f;
#pragma unroll
        for (int j = 0; j < 8; ++j)
            colSum += invz[j] * __expf(mrow[j] - (p2 - 2.0f * c[tt][j]));
        colSum += __shfl_xor(colSum, 16, 32);   // combine 8 rows from each half
        if (lane < 16) {
            const int colW = (tt >> 3) * HCOLS + wc * 128 + (tt & 7) * 16 + lane;
            atomicAdd(&sSoft[colW], colSum);
        }
    }
    __syncthreads();

    // ---- flush soft partials, gather winning prototypes, fused MSE --------
    for (int i = tid; i < NPROTO; i += THREADS) atomicAdd(&gSoft[i], sSoft[i]);

    {
        const int r  = tid >> 3;            // row 0..63 (8 threads per row)
        const int d0 = (tid & 7) * 64;      // 64 elems each
        const int pidx = sRowIdx[r];
        const float4* gp = reinterpret_cast<const float4*>(protos + (size_t)pidx * DIM + d0);
        float4* go = reinterpret_cast<float4*>(outQ + (size_t)(n0 + r) * DIM + d0);
        float mseAcc = 0.f;
#pragma unroll
        for (int v = 0; v < 16; ++v) {
            float4 p = gp[v];
            const int base = r * DIM + d0 + v * 4;
            float x0 = (float)sAhi[base + 0] + (float)sAlo[base + 0];
            float x1 = (float)sAhi[base + 1] + (float)sAlo[base + 1];
            float x2 = (float)sAhi[base + 2] + (float)sAlo[base + 2];
            float x3 = (float)sAhi[base + 3] + (float)sAlo[base + 3];
            float d0f = p.x - x0, d1f = p.y - x1, d2f = p.z - x2, d3f = p.w - x3;
            mseAcc += d0f * d0f + d1f * d1f + d2f * d2f + d3f * d3f;
            go[v] = p;
        }
        atomicAdd(sMse, mseAcc);
    }
    __syncthreads();
    if (tid == 0) atomicAdd(gMse, sMse[0]);
}

// ---------------------------------------------------------------------------
// Kernel C: finalize loss = 1.25*mse + 0.01*entropy(normalized soft)
// ---------------------------------------------------------------------------
__global__ void vq_fin(const float* __restrict__ gSoft, const float* __restrict__ gMse,
                       float* __restrict__ outLoss) {
    __shared__ float red[256];
    __shared__ float sInv;
    const int tid = threadIdx.x;
    float a = 0.f;
    for (int i = tid; i < NPROTO; i += 256)
        a += gSoft[i] * (1.0f / (float)N_TOK) + 1e-6f;
    red[tid] = a; __syncthreads();
    for (int s = 128; s > 0; s >>= 1) {
        if (tid < s) red[tid] += red[tid + s];
        __syncthreads();
    }
    if (tid == 0) sInv = 1.0f / red[0];
    __syncthreads();
    const float inv = sInv;
    float e = 0.f;
    for (int i = tid; i < NPROTO; i += 256) {
        float s = (gSoft[i] * (1.0f / (float)N_TOK) + 1e-6f) * inv;
        e += -s * __logf(s);
    }
    red[tid] = e; __syncthreads();
    for (int s = 128; s > 0; s >>= 1) {
        if (tid < s) red[tid] += red[tid + s];
        __syncthreads();
    }
    if (tid == 0) {
        float mse = gMse[0] * (1.0f / ((float)N_TOK * (float)DIM));
        outLoss[0] = 1.25f * mse + 0.01f * red[0];
    }
}

// ---------------------------------------------------------------------------
extern "C" void kernel_launch(void* const* d_in, const int* in_sizes, int n_in,
                              void* d_out, int out_size, void* d_ws, size_t ws_size,
                              hipStream_t stream) {
    (void)in_sizes; (void)n_in; (void)out_size; (void)ws_size;
    const float* latents = (const float*)d_in[0];
    const float* protos  = (const float*)d_in[1];
    float* out = (float*)d_out;                     // [N*D] quantized + [1] loss

    char* ws = (char*)d_ws;                         // ~2.01 MB used
    __bf16* pHi  = (__bf16*)(ws);                   // 1 MB
    __bf16* pLo  = (__bf16*)(ws + 1048576);         // 1 MB
    float*  pn2  = (float*)(ws + 2097152);          // 4 KB
    float*  gSoft = (float*)(ws + 2097152 + 4096);  // 4 KB
    float*  gMse  = (float*)(ws + 2097152 + 8192);  // 4 B

    vq_prep<<<dim3(NPROTO), dim3(256), 0, stream>>>(protos, pHi, pLo, pn2, gSoft, gMse);
    vq_main<<<dim3(N_TOK / BM), dim3(THREADS), LDS_BYTES, stream>>>(
        latents, protos, pHi, pLo, pn2, gSoft, gMse, out);
    vq_fin<<<dim3(1), dim3(256), 0, stream>>>(gSoft, gMse, out + (size_t)N_TOK * DIM);
}
